// VectorQuantizer_592705487401
// MI455X (gfx1250) — matrix operand order repaired and checked by reading, not compiled, not run on hardware
//
#include <hip/hip_runtime.h>
#include <hip/hip_bf16.h>
#include <stdint.h>

#define N_TOK 8192
#define N_CODE 8192
#define DIM 512
#define TN 128                      // tokens per block (8 waves x 16 rows)
#define SEGS 8                      // codebook segments (for occupancy)
#define SEG_CODES (N_CODE / SEGS)   // 1024 codes per segment
#define CHUNK 32                    // codes per staged chunk (2 x 16-wide tiles)
#define NCHUNK (SEG_CODES / CHUNK)  // 32
#define TILE_ELEMS (16 * 32 * 16)   // one 16-code fragment tile in LDS (bf16 elems)

typedef __attribute__((ext_vector_type(16))) __bf16 v16bf;
typedef __attribute__((ext_vector_type(4)))  __bf16 v4bf;
typedef __attribute__((ext_vector_type(8)))  float  v8f;

__device__ __forceinline__ unsigned int f32_ordered(float f) {
  unsigned int u = __float_as_uint(f);
  return (u & 0x80000000u) ? ~u : (u | 0x80000000u);
}

// ---------------------------------------------------------------- init
__global__ void vq_init(unsigned long long* __restrict__ packed,
                        float* __restrict__ loss) {
  int i = blockIdx.x * blockDim.x + threadIdx.x;
  if (i < N_TOK) packed[i] = 0xFFFFFFFFFFFFFFFFULL;
  if (i == 0) *loss = 0.0f;
}

// ---------------------------------------------------------------- ||c||^2
__global__ void vq_cnorm(const float* __restrict__ cb,
                         float* __restrict__ cnorm) {
  int wave = threadIdx.x >> 5;
  int lane = threadIdx.x & 31;
  int row  = blockIdx.x * 8 + wave;
  const float* r = cb + (size_t)row * DIM;
  float s = 0.0f;
#pragma unroll
  for (int i = 0; i < DIM / 32; ++i) { float v = r[lane + 32 * i]; s += v * v; }
#pragma unroll
  for (int off = 16; off; off >>= 1) s += __shfl_xor(s, off, 32);
  if (lane == 0) cnorm[row] = s;
}

// ---------------------------------------------------------------- fused GEMM + argmin
__global__ __launch_bounds__(256, 1) void vq_argmin(
    const float* __restrict__ flat, const float* __restrict__ cb,
    const float* __restrict__ cnorm, unsigned long long* __restrict__ packed) {
  // Double-buffered B staging: 2 buffers x 2 tiles x 16KB = 64 KB
  __shared__ alignas(32) __bf16 sB[2][2 * TILE_ELEMS];

  const int tid  = threadIdx.x;
  const int wave = tid >> 5;
  const int lane = tid & 31;
  const int l15  = lane & 15;
  const int hi   = lane >> 4;
  const int tokenBase = (blockIdx.x & 63) * TN;
  const int segBase   = (blockIdx.x >> 6) * SEG_CODES;

  // A fragments: 16 token rows per wave, resident in VGPRs as bf16.
  // 16-bit A layout used here (same as B): K(e) = e + 16*hi
  // -> groups of 4 elements are K-contiguous, so float4 loads work.
  const int row = tokenBase + wave * 16 + l15;
  const float* rp = flat + (size_t)row * DIM;
  v16bf a[16];
#pragma unroll
  for (int ks = 0; ks < 16; ++ks) {
#pragma unroll
    for (int q = 0; q < 4; ++q) {
      int kb = ks * 32 + hi * 16 + q * 4;
      float4 f = *(const float4*)(rp + kb);
      a[ks][4 * q + 0] = (__bf16)f.x;
      a[ks][4 * q + 1] = (__bf16)f.y;
      a[ks][4 * q + 2] = (__bf16)f.z;
      a[ks][4 * q + 3] = (__bf16)f.w;
    }
  }

  // Cooperative stage: CHUNK codebook rows -> bf16 B-fragment layout.
  // B layout: lane = n + 16*hi, K(e) = e + 16*hi. 4 consecutive K share a lane,
  // so one float4 global load becomes one 8-byte ds_store.
  auto stage = [&](int buf, int cb0) {
#pragma unroll
    for (int it = 0; it < (CHUNK * DIM) / (256 * 4); ++it) {  // 16 iters
      int i = (tid + it * 256) * 4;
      int c = i >> 9;            // 0..31
      int k = i & 511;           // multiple of 4
      int ks = k >> 5, kk = k & 31;
      int bhi = kk >> 4;
      int e   = kk & 15;         // multiple of 4
      int bl  = (c & 15) | (bhi << 4);
      int tile = c >> 4;
      float4 f = *(const float4*)(cb + (size_t)(cb0 + c) * DIM + k);
      v4bf v;
      v[0] = (__bf16)f.x; v[1] = (__bf16)f.y;
      v[2] = (__bf16)f.z; v[3] = (__bf16)f.w;
      *(v4bf*)&sB[buf][(tile * 16 + ks) * 32 * 16 + (size_t)bl * 16 + e] = v;
    }
  };

  unsigned long long run[8];
#pragma unroll
  for (int j = 0; j < 8; ++j) run[j] = 0xFFFFFFFFFFFFFFFFULL;

  stage(0, segBase);

  for (int chunk = 0; chunk < NCHUNK; ++chunk) {
    const int cBase = segBase + chunk * CHUNK;
    __syncthreads();  // staged buffer (chunk&1) visible; previous reads done

    // Stage next chunk into the other buffer while this one computes.
    if (chunk + 1 < NCHUNK) stage((chunk + 1) & 1, cBase + CHUNK);

    // Hint-prefetch chunk+2 (2KB rows, 256B per thread).
    if (cBase + 2 * CHUNK < segBase + SEG_CODES) {
      const char* p = (const char*)(cb + (size_t)(cBase + 2 * CHUNK + (tid & 31)) * DIM)
                      + (tid >> 5) * 256;
      __builtin_prefetch(p, 0, 1);
    }

    // Two independent accumulator tiles -> 2 WMMA chains, 4 ds_loads per step.
    const __bf16* bufp = sB[chunk & 1];
    v8f acc0 = {}, acc1 = {};
#pragma unroll
    for (int ks = 0; ks < 16; ++ks) {
      const __bf16* p = bufp + ((size_t)ks * 32 + lane) * 16;
      v16bf b0 = *(const v16bf*)p;
      v16bf b1 = *(const v16bf*)(p + TILE_ELEMS);
      acc0 = __builtin_amdgcn_wmma_f32_16x16x32_bf16(
          false, a[ks], false, b0, (short)0, acc0, false, false);
      acc1 = __builtin_amdgcn_wmma_f32_16x16x32_bf16(
          false, a[ks], false, b1, (short)0, acc1, false, false);
    }

    // dist = ||c||^2 - 2 x.c  (||x||^2 constant per row -> dropped)
    float cn0 = cnorm[cBase + l15];
    float cn1 = cnorm[cBase + 16 + l15];
    unsigned long long code0 = (unsigned long long)(unsigned)(cBase + l15);
    unsigned long long code1 = code0 + 16;
#pragma unroll
    for (int j = 0; j < 8; ++j) {
      float d0 = __builtin_fmaf(-2.0f, acc0[j], cn0);
      float d1 = __builtin_fmaf(-2.0f, acc1[j], cn1);
      unsigned long long p0 = ((unsigned long long)f32_ordered(d0) << 32) | code0;
      unsigned long long p1 = ((unsigned long long)f32_ordered(d1) << 32) | code1;
      unsigned long long pk = p0 < p1 ? p0 : p1;
      run[j] = pk < run[j] ? pk : run[j];
    }
  }

  // Reduce across the 16 lanes of each half (C layout: VGPR j -> M=j+8*hi, lane -> N).
#pragma unroll
  for (int j = 0; j < 8; ++j) {
    unsigned long long v = run[j];
#pragma unroll
    for (int off = 8; off; off >>= 1) {
      unsigned long long o = __shfl_xor(v, off, 16);
      v = o < v ? o : v;
    }
    if (l15 == 0) {
      int r = tokenBase + wave * 16 + j + 8 * hi;
      atomicMin(&packed[r], v);
    }
  }
}

// ---------------------------------------------------------------- gather + fused loss
__global__ void vq_gather(const float* __restrict__ flat,
                          const float* __restrict__ cb,
                          const unsigned long long* __restrict__ packed,
                          float* __restrict__ out, float* __restrict__ loss) {
  __shared__ float warpSum[4];
  int n   = blockIdx.x;
  int tid = threadIdx.x;  // 128 threads, 4 floats each
  unsigned int idx = (unsigned int)(packed[n] & 0xFFFFFFFFu);
  const float4* q = (const float4*)(cb + (size_t)idx * DIM);
  const float4* f = (const float4*)(flat + (size_t)n * DIM);
  float4* o = (float4*)(out + (size_t)n * DIM);
  float4 qv = q[tid];
  float4 fv = f[tid];
  o[tid] = qv;  // straight-through forward value == q
  float dx = qv.x - fv.x, dy = qv.y - fv.y, dz = qv.z - fv.z, dw = qv.w - fv.w;
  float s = dx * dx + dy * dy + dz * dz + dw * dw;
#pragma unroll
  for (int off = 16; off; off >>= 1) s += __shfl_xor(s, off, 32);
  if ((tid & 31) == 0) warpSum[tid >> 5] = s;
  __syncthreads();
  if (tid == 0) {
    atomicAdd(loss, warpSum[0] + warpSum[1] + warpSum[2] + warpSum[3]);
    out[(size_t)N_TOK * DIM + 1 + n] = (float)idx;  // idx output region
  }
}

// ---------------------------------------------------------------- final scalar
__global__ void vq_loss_final(const float* __restrict__ loss,
                              float* __restrict__ out) {
  // commit_loss == codebook_loss numerically -> (1 + 0.25) * mean
  out[(size_t)N_TOK * DIM] = 1.25f * loss[0] / (float)((size_t)N_TOK * DIM);
}

extern "C" void kernel_launch(void* const* d_in, const int* in_sizes, int n_in,
                              void* d_out, int out_size, void* d_ws, size_t ws_size,
                              hipStream_t stream) {
  (void)in_sizes; (void)n_in; (void)out_size; (void)ws_size;
  const float* tokens = (const float*)d_in[0];   // [8192,512] f32
  const float* cb     = (const float*)d_in[1];   // [8192,512] f32
  float* out = (float*)d_out;

  float*              cnorm  = (float*)d_ws;                                    // 32 KB
  unsigned long long* packed = (unsigned long long*)((char*)d_ws + 32 * 1024);  // 64 KB
  float*              loss   = (float*)((char*)d_ws + 96 * 1024);               // 4 B

  vq_init      <<<N_TOK / 256, 256, 0, stream>>>(packed, loss);
  vq_cnorm     <<<N_CODE / 8, 256, 0, stream>>>(cb, cnorm);
  vq_argmin    <<<(N_TOK / TN) * SEGS, 256, 0, stream>>>(tokens, cb, cnorm, packed);
  vq_gather    <<<N_TOK, 128, 0, stream>>>(tokens, cb, packed, out, loss);
  vq_loss_final<<<1, 1, 0, stream>>>(loss, out);
}
